// ASTGCNBlock_56908316672451
// MI455X (gfx1250) — matrix operand order repaired and checked by reading, not compiled, not run on hardware
//
#include <hip/hip_runtime.h>
#include <hip/hip_bf16.h>

typedef __attribute__((ext_vector_type(16))) _Float16 v16h;
typedef __attribute__((ext_vector_type(8)))  _Float16 v8h;
typedef __attribute__((ext_vector_type(4)))  _Float16 v4h;
typedef __attribute__((ext_vector_type(8)))  float    v8f;

#define NN_NODES 4000
#define LN_EPS 1e-5f

// ---------------------------------------------------------------------------
// f32 -> f16 conversion of Vs and laplacian (16M elements each)
// ---------------------------------------------------------------------------
__global__ __launch_bounds__(256) void k_cvt(const float* __restrict__ Vs,
                                             const float* __restrict__ lap,
                                             _Float16* __restrict__ Vs16,
                                             _Float16* __restrict__ lap16) {
  size_t base = ((size_t)blockIdx.x * 256 + threadIdx.x) * 4;
  #pragma unroll
  for (int k = 0; k < 4; k++) {
    Vs16[base + k]  = (_Float16)Vs[base + k];
    lap16[base + k] = (_Float16)lap[base + k];
  }
}

// ---------------------------------------------------------------------------
// Temporal attention partial reductions (deterministic two-stage)
// grid = B*250 blocks, 16 n per block
// ---------------------------------------------------------------------------
__global__ __launch_bounds__(256) void k_temporal(const float* __restrict__ x,
                                                  const float* __restrict__ U1,
                                                  const float* __restrict__ U2,
                                                  const float* __restrict__ U3,
                                                  float* __restrict__ part_lhs,
                                                  float* __restrict__ part_rhs) {
  __shared__ float redL[16][17];
  __shared__ float redR[16][17];
  int b  = blockIdx.x / 250;
  int n0 = (blockIdx.x % 250) * 16;
  int ni = threadIdx.x >> 4, tt = threadIdx.x & 15;
  int n  = n0 + ni;
  const float* xp = x + (size_t)(b * NN_NODES + n) * 256 + tt;
  float sx = 0.f;
  #pragma unroll
  for (int f = 0; f < 16; f++) sx += xp[f * 16];
  float u = 0.f;
  #pragma unroll
  for (int g = 0; g < 16; g++) u += U2[g * NN_NODES + n] * U3[g];
  redL[ni][tt] = sx * U1[n];
  redR[ni][tt] = sx * u;
  __syncthreads();
  if (ni == 0) {
    float sL = 0.f, sR = 0.f;
    #pragma unroll
    for (int k = 0; k < 16; k++) { sL += redL[k][tt]; sR += redR[k][tt]; }
    part_lhs[blockIdx.x * 16 + tt] = sL;
    part_rhs[blockIdx.x * 16 + tt] = sR;
  }
}

// ---------------------------------------------------------------------------
// Finish t_lhs/t_rhs reduction, compute E[b,t,r] (sigmoid + Ve matmul + softmax)
// ---------------------------------------------------------------------------
__global__ __launch_bounds__(256) void k_E(const float* __restrict__ part_lhs,
                                           const float* __restrict__ part_rhs,
                                           const float* __restrict__ be,
                                           const float* __restrict__ Ve,
                                           float* __restrict__ E) {
  __shared__ float lhs_s[64], rhs_s[64];
  __shared__ float sig[16][16];
  __shared__ float ep[16][16];
  int tid = threadIdx.x;
  if (tid < 64) {
    int b = tid >> 4, t = tid & 15;
    float sL = 0.f, sR = 0.f;
    for (int k = 0; k < 250; k++) {
      sL += part_lhs[(b * 250 + k) * 16 + t];
      sR += part_rhs[(b * 250 + k) * 16 + t];
    }
    lhs_s[tid] = sL; rhs_s[tid] = sR;
  }
  __syncthreads();
  int i = tid >> 4, j = tid & 15;
  for (int b = 0; b < 4; b++) {
    float v = lhs_s[b * 16 + i] * rhs_s[b * 16 + j] + be[i * 16 + j];
    sig[i][j] = 1.f / (1.f + __expf(-v));
    __syncthreads();
    float s = 0.f;
    #pragma unroll
    for (int k = 0; k < 16; k++) s += Ve[i * 16 + k] * sig[k][j];
    ep[i][j] = s;
    __syncthreads();
    float mx = ep[i][0];
    #pragma unroll
    for (int k = 1; k < 16; k++) mx = fmaxf(mx, ep[i][k]);
    float den = 0.f;
    #pragma unroll
    for (int k = 0; k < 16; k++) den += __expf(ep[i][k] - mx);
    E[(b * 16 + i) * 16 + j] = __expf(ep[i][j] - mx) / den;
    __syncthreads();
  }
}

// ---------------------------------------------------------------------------
// Per (b,n): x_TAt (f16), s_lhs, s_rhs, y_base = time_out + time_b + residual
// ---------------------------------------------------------------------------
__global__ __launch_bounds__(256) void k_pernode(const float* __restrict__ x,
                                                 const float* __restrict__ E,
                                                 const float* __restrict__ W1,
                                                 const float* __restrict__ W2,
                                                 const float* __restrict__ W3,
                                                 const float* __restrict__ time_w,
                                                 const float* __restrict__ time_b,
                                                 _Float16* __restrict__ xTAt,
                                                 float* __restrict__ s_lhs,
                                                 float* __restrict__ s_rhs,
                                                 float* __restrict__ y_base) {
  __shared__ float xs[16][17];
  __shared__ float Eb[16][16];
  __shared__ float xtl[16];
  int bn = blockIdx.x;
  int b = bn / NN_NODES;
  int f = threadIdx.x >> 4, tt = threadIdx.x & 15;
  xs[f][tt] = x[(size_t)bn * 256 + f * 16 + tt];
  Eb[f][tt] = E[b * 256 + f * 16 + tt];  // Eb[t][s]
  __syncthreads();
  float acc = 0.f;
  #pragma unroll
  for (int s = 0; s < 16; s++) acc += Eb[tt][s] * xs[f][s];
  xTAt[(size_t)bn * 256 + f * 16 + tt] = (_Float16)acc;
  if (tt == 0) {
    float s = 0.f;
    #pragma unroll
    for (int t2 = 0; t2 < 16; t2++) s += xs[f][t2] * W1[t2];
    s_lhs[bn * 16 + f] = s;
  } else if (tt == 1) {
    float s = 0.f;
    #pragma unroll
    for (int t2 = 0; t2 < 16; t2++) s += xs[f][t2] * W3[t2];
    xtl[f] = s;
  }
  __syncthreads();
  if (threadIdx.x < 16) {
    int g = threadIdx.x;
    float s = 0.f;
    #pragma unroll
    for (int k2 = 0; k2 < 16; k2++) s += xtl[k2] * W2[g * 16 + k2];
    s_rhs[bn * 16 + g] = s;
    float to = 0.f;
    for (int c = 0; c < 16; c++)
      #pragma unroll
      for (int t2 = 0; t2 < 16; t2++)
        to += xs[c][t2] * time_w[g * 256 + c * 16 + t2];
    y_base[bn * 16 + g] = to + time_b[g] + xs[g][15];
  }
}

// ---------------------------------------------------------------------------
// P[m][k] = sigmoid(dot(s_lhs[b,m,:], s_rhs[b,k,:]) + bs[m,k]) -> f16
// ---------------------------------------------------------------------------
__global__ __launch_bounds__(256) void k_buildP(const float* __restrict__ s_lhs_b,
                                                const float* __restrict__ s_rhs_b,
                                                const float* __restrict__ bs,
                                                _Float16* __restrict__ P) {
  __shared__ float Ls[64][16];
  __shared__ float Rs[64][16];
  int m0 = blockIdx.y * 64;
  int k0 = blockIdx.x * 64;
  int t = threadIdx.x;
  {
    int row = t >> 2;
    int q = (t & 3) * 4;
    #pragma unroll
    for (int k = 0; k < 4; k++) {
      Ls[row][q + k] = (m0 + row < NN_NODES) ? s_lhs_b[(m0 + row) * 16 + q + k] : 0.f;
      Rs[row][q + k] = (k0 + row < NN_NODES) ? s_rhs_b[(k0 + row) * 16 + q + k] : 0.f;
    }
  }
  __syncthreads();
  int c = t & 63;
  int r0 = (t >> 6) * 16;
  float rc[16];
  #pragma unroll
  for (int k = 0; k < 16; k++) rc[k] = Rs[c][k];
  if (k0 + c < NN_NODES) {
    for (int r = 0; r < 16; r++) {
      int row = m0 + r0 + r;
      if (row < NN_NODES) {
        float d = 0.f;
        #pragma unroll
        for (int k = 0; k < 16; k++) d += Ls[r0 + r][k] * rc[k];
        size_t idx = (size_t)row * NN_NODES + k0 + c;
        float v = d + bs[idx];
        P[idx] = (_Float16)(1.f / (1.f + __expf(-v)));
      }
    }
  }
}

// ---------------------------------------------------------------------------
// WMMA GEMM: C[MxNcol] = A[MxK](f16,row-major) x B[KxNcol](f16,row-major)
// f32 accumulate.  mode 0: store f32   mode 1: store f16
// mode 2: store f16 of (2*acc - aux[idx])   (Chebyshev recurrence epilogue)
// block tile 128x128, 8 waves (2x4), each wave 4x2 16x16 tiles, k-step 32.
// Register-staged double-buffered LDS pipeline: one barrier per k-step,
// global fetch of step s+1 overlaps the 8 WMMAs of step s.
// ---------------------------------------------------------------------------
__global__ __launch_bounds__(256) void gemm_wmma_f16(const _Float16* __restrict__ A,
                                                     const _Float16* __restrict__ B,
                                                     float* __restrict__ Cf,
                                                     _Float16* __restrict__ Ch,
                                                     const _Float16* __restrict__ aux,
                                                     int M, int K, int Ncol, int mode) {
  // padded stride 40 halves (80B = 5*16B: keeps b128 alignment, spreads banks)
  __shared__ _Float16 As[2][128 * 40];
  __shared__ _Float16 Bs[2][128 * 40];   // transposed: Bs[p][c*40 + k]

  const int tid  = threadIdx.x;
  const int lane = tid & 31;
  const int wave = tid >> 5;
  const int wm = wave >> 2;   // 0..1
  const int wn = wave & 3;    // 0..3
  const int ll = lane & 15;
  const int lh = lane >> 4;

  const int m0 = blockIdx.y * 128;
  const int n0 = blockIdx.x * 128;

  v8f acc[4][2];
  #pragma unroll
  for (int i = 0; i < 4; i++)
    #pragma unroll
    for (int j = 0; j < 2; j++)
      #pragma unroll
      for (int v = 0; v < 8; v++) acc[i][j][v] = 0.f;

  // loader mappings (loop-invariant)
  const int ar  = tid >> 1;         // A: row 0..127
  const int ap  = (tid & 1) * 16;   // A: half-row chunk (halves)
  const int bc0 = (tid & 31) * 4;   // B: col group
  const int bk0 = (tid >> 5) * 4;   // B: k group

  const bool arow_ok = (m0 + ar) < M;
  const bool bfull   = (n0 + bc0 + 3) < Ncol;
  bool bok[4];
  #pragma unroll
  for (int cc = 0; cc < 4; cc++) bok[cc] = (n0 + bc0 + cc) < Ncol;

  const _Float16* Aptr = A + (size_t)(m0 + ar) * K + ap;
  const _Float16* Bptr = B + (size_t)bk0 * Ncol + n0 + bc0;

  const int nstep = K >> 5;

  v8h ra0, ra1;   // staged A regs
  v4h rb[4];      // staged B regs: rb[kk][cc]

  auto load_glb = [&](int s) {
    const int ks = s << 5;
    if (arow_ok) {
      const _Float16* src = Aptr + ks;
      ra0 = *(const v8h*)(src);
      ra1 = *(const v8h*)(src + 8);
    } else {
      #pragma unroll
      for (int q = 0; q < 8; q++) { ra0[q] = (_Float16)0.f; ra1[q] = (_Float16)0.f; }
    }
    const _Float16* bsrc = Bptr + (size_t)ks * Ncol;
    if (bfull) {
      #pragma unroll
      for (int kk = 0; kk < 4; kk++) rb[kk] = *(const v4h*)(bsrc + (size_t)kk * Ncol);
    } else {
      #pragma unroll
      for (int kk = 0; kk < 4; kk++)
        #pragma unroll
        for (int cc = 0; cc < 4; cc++)
          rb[kk][cc] = bok[cc] ? bsrc[(size_t)kk * Ncol + cc] : (_Float16)0.f;
    }
  };
  auto store_lds = [&](int p) {
    *(v8h*)(&As[p][ar * 40 + ap])     = ra0;
    *(v8h*)(&As[p][ar * 40 + ap + 8]) = ra1;
    #pragma unroll
    for (int cc = 0; cc < 4; cc++) {
      v4h tv;
      #pragma unroll
      for (int kk = 0; kk < 4; kk++) tv[kk] = rb[kk][cc];
      *(v4h*)(&Bs[p][(bc0 + cc) * 40 + bk0]) = tv;
    }
  };

  load_glb(0);
  store_lds(0);

  for (int s = 0; s < nstep; s++) {
    const int p = s & 1;
    if (s + 1 < nstep) load_glb(s + 1);     // overlap next fetch with compute
    __syncthreads();                         // buf p ready for all waves
    // ---- fragments + WMMA from buffer p
    union Frag { v16h v; v8h h[2]; };
    Frag af[4], bf[2];
    #pragma unroll
    for (int i = 0; i < 4; i++) {
      int row = wm * 64 + i * 16 + ll;
      const _Float16* ptr = &As[p][row * 40];
      af[i].h[0] = *(const v8h*)(ptr + lh * 8);       // k = 8h .. 8h+7
      af[i].h[1] = *(const v8h*)(ptr + 16 + lh * 8);  // k = 16+8h .. 16+8h+7
    }
    #pragma unroll
    for (int j = 0; j < 2; j++) {
      int col = wn * 32 + j * 16 + ll;
      const _Float16* ptr = &Bs[p][col * 40];
      bf[j].h[0] = *(const v8h*)(ptr + lh * 8);       // k = 8h .. 8h+7
      bf[j].h[1] = *(const v8h*)(ptr + 16 + lh * 8);  // k = 16+8h .. 16+8h+7
    }
    #pragma unroll
    for (int i = 0; i < 4; i++)
      #pragma unroll
      for (int j = 0; j < 2; j++)
        acc[i][j] = __builtin_amdgcn_wmma_f32_16x16x32_f16(
            false, af[i].v, false, bf[j].v, (short)0, acc[i][j], false, false);
    // tail: publish tile s+1 into the other buffer (safe: all waves finished
    // reading buf 1-p before the barrier above)
    if (s + 1 < nstep) store_lds(1 - p);
  }

  // ---- epilogue (uniform interior fast path)
  const bool interior = (m0 + 128 <= M) && (n0 + 128 <= Ncol);
  #pragma unroll
  for (int i = 0; i < 4; i++) {
    #pragma unroll
    for (int j = 0; j < 2; j++) {
      int col = n0 + wn * 32 + j * 16 + ll;
      int rbase = m0 + wm * 64 + i * 16 + lh * 8;
      if (interior) {
        #pragma unroll
        for (int v = 0; v < 8; v++) {
          size_t idx = (size_t)(rbase + v) * Ncol + col;
          float val = acc[i][j][v];
          if (mode == 0)      Cf[idx] = val;
          else if (mode == 1) Ch[idx] = (_Float16)val;
          else                Ch[idx] = (_Float16)(2.0f * val - (float)aux[idx]);
        }
      } else if (col < Ncol) {
        #pragma unroll
        for (int v = 0; v < 8; v++) {
          if (rbase + v < M) {
            size_t idx = (size_t)(rbase + v) * Ncol + col;
            float val = acc[i][j][v];
            if (mode == 0)      Cf[idx] = val;
            else if (mode == 1) Ch[idx] = (_Float16)val;
            else                Ch[idx] = (_Float16)(2.0f * val - (float)aux[idx]);
          }
        }
      }
    }
  }
}

// ---------------------------------------------------------------------------
// Row softmax: Spre (f32, 4000 per row) -> S (f16)
// ---------------------------------------------------------------------------
__global__ __launch_bounds__(256) void k_softmax_row(const float* __restrict__ Spre,
                                                     _Float16* __restrict__ S) {
  __shared__ float red[256];
  int row = blockIdx.x;
  const float* p = Spre + (size_t)row * NN_NODES;
  float mx = -3.4e38f;
  for (int i = threadIdx.x; i < NN_NODES; i += 256) mx = fmaxf(mx, p[i]);
  red[threadIdx.x] = mx;
  __syncthreads();
  for (int s = 128; s > 0; s >>= 1) {
    if (threadIdx.x < s) red[threadIdx.x] = fmaxf(red[threadIdx.x], red[threadIdx.x + s]);
    __syncthreads();
  }
  mx = red[0];
  __syncthreads();
  float sum = 0.f;
  for (int i = threadIdx.x; i < NN_NODES; i += 256) sum += __expf(p[i] - mx);
  red[threadIdx.x] = sum;
  __syncthreads();
  for (int s = 128; s > 0; s >>= 1) {
    if (threadIdx.x < s) red[threadIdx.x] += red[threadIdx.x + s];
    __syncthreads();
  }
  float inv = 1.f / red[0];
  _Float16* q = S + (size_t)row * NN_NODES;
  for (int i = threadIdx.x; i < NN_NODES; i += 256)
    q[i] = (_Float16)(__expf(p[i] - mx) * inv);
}

// ---------------------------------------------------------------------------
// Chebyshev projection + mean over t + y_base; per-block layernorm partials
// ---------------------------------------------------------------------------
__global__ __launch_bounds__(256) void k_cheb(const _Float16* __restrict__ Tx0,
                                              const _Float16* __restrict__ Tx1,
                                              const _Float16* __restrict__ Tx2,
                                              const float* __restrict__ cheb_w,
                                              const float* __restrict__ y_base,
                                              float* __restrict__ y_pre,
                                              float* __restrict__ part_st) {
  __shared__ float t0[16][17], t1[16][17], t2[16][17];
  __shared__ float red[16][17];
  __shared__ float osum[16];
  int bn = blockIdx.x;
  int f = threadIdx.x >> 4, tt = threadIdx.x & 15;
  size_t base = (size_t)bn * 256 + f * 16 + tt;
  t0[f][tt] = (float)Tx0[base];
  t1[f][tt] = (float)Tx1[base];
  t2[f][tt] = (float)Tx2[base];
  __syncthreads();
  int o = f;
  float v = 0.f;
  #pragma unroll
  for (int ff = 0; ff < 16; ff++)
    v += t0[ff][tt] * cheb_w[ff * 16 + o]
       + t1[ff][tt] * cheb_w[256 + ff * 16 + o]
       + t2[ff][tt] * cheb_w[512 + ff * 16 + o];
  red[o][tt] = v;
  __syncthreads();
  if (tt == 0) {
    float s = 0.f;
    #pragma unroll
    for (int k = 0; k < 16; k++) s += red[o][k];
    float y = y_base[bn * 16 + o] + s * (1.f / 16.f);
    y_pre[bn * 16 + o] = y;
    osum[o] = y;
  }
  __syncthreads();
  if (threadIdx.x == 0) {
    float s = 0.f, s2 = 0.f;
    #pragma unroll
    for (int k = 0; k < 16; k++) { s += osum[k]; s2 += osum[k] * osum[k]; }
    part_st[bn * 2]     = s;
    part_st[bn * 2 + 1] = s2;
  }
}

// ---------------------------------------------------------------------------
// Reduce layernorm partials (deterministic), one block per b
// ---------------------------------------------------------------------------
__global__ __launch_bounds__(256) void k_stats(const float* __restrict__ part,
                                               float* __restrict__ stats) {
  __shared__ float r0[256], r1[256];
  int b = blockIdx.x;
  float s0 = 0.f, s1 = 0.f;
  for (int i = threadIdx.x; i < NN_NODES; i += 256) {
    s0 += part[(b * NN_NODES + i) * 2];
    s1 += part[(b * NN_NODES + i) * 2 + 1];
  }
  r0[threadIdx.x] = s0; r1[threadIdx.x] = s1;
  __syncthreads();
  for (int s = 128; s > 0; s >>= 1) {
    if (threadIdx.x < s) { r0[threadIdx.x] += r0[threadIdx.x + s]; r1[threadIdx.x] += r1[threadIdx.x + s]; }
    __syncthreads();
  }
  if (threadIdx.x == 0) { stats[b * 2] = r0[0]; stats[b * 2 + 1] = r1[0]; }
}

// ---------------------------------------------------------------------------
// Layernorm + relu -> output
// ---------------------------------------------------------------------------
__global__ __launch_bounds__(256) void k_norm(const float* __restrict__ y_pre,
                                              const float* __restrict__ stats,
                                              const float* __restrict__ ln_g,
                                              const float* __restrict__ ln_b,
                                              float* __restrict__ out) {
  int idx = blockIdx.x * 256 + threadIdx.x;  // 0..255999
  int b  = idx / 64000;
  int no = idx % 64000;
  const float cnt = 64000.f;
  float mean = stats[b * 2] / cnt;
  float var  = stats[b * 2 + 1] / cnt - mean * mean;
  float y = (y_pre[idx] - mean) * rsqrtf(var + LN_EPS) * ln_g[no] + ln_b[no];
  out[idx] = fmaxf(y, 0.f);
}

// ---------------------------------------------------------------------------
extern "C" void kernel_launch(void* const* d_in, const int* in_sizes, int n_in,
                              void* d_out, int out_size, void* d_ws, size_t ws_size,
                              hipStream_t stream) {
  const float* x    = (const float*)d_in[0];
  const float* lap  = (const float*)d_in[1];
  const float* W1   = (const float*)d_in[2];
  const float* W2   = (const float*)d_in[3];
  const float* W3   = (const float*)d_in[4];
  const float* bs   = (const float*)d_in[5];
  const float* Vs   = (const float*)d_in[6];
  const float* U1   = (const float*)d_in[7];
  const float* U2   = (const float*)d_in[8];
  const float* U3   = (const float*)d_in[9];
  const float* be   = (const float*)d_in[10];
  const float* Ve   = (const float*)d_in[11];
  const float* cheb = (const float*)d_in[12];
  const float* tw   = (const float*)d_in[13];
  const float* tb   = (const float*)d_in[14];
  const float* ln_g = (const float*)d_in[15];
  const float* ln_b = (const float*)d_in[16];
  float* out = (float*)d_out;

  char* w = (char*)d_ws;
  _Float16* Vs16  = (_Float16*)(w + 0ull);
  _Float16* lap16 = (_Float16*)(w + 32000000ull);
  _Float16* P16   = (_Float16*)(w + 64000000ull);   // per-b reuse
  _Float16* S16   = (_Float16*)(w + 96000000ull);   // per-b reuse
  float*    Spre  = (float*)   (w + 128000000ull);  // per-b reuse (f32 logits)
  _Float16* xTAt  = (_Float16*)(w + 192000000ull);
  _Float16* Tx0   = (_Float16*)(w + 200192000ull);
  _Float16* Tx1   = (_Float16*)(w + 208384000ull);
  _Float16* Tx2   = (_Float16*)(w + 216576000ull);
  float* s_lhs    = (float*)(w + 224768000ull);
  float* s_rhs    = (float*)(w + 225792000ull);
  float* y_base   = (float*)(w + 226816000ull);
  float* y_pre    = (float*)(w + 227840000ull);
  float* E        = (float*)(w + 228864000ull);
  float* part_lhs = (float*)(w + 228868096ull);
  float* part_rhs = (float*)(w + 228932096ull);
  float* part_st  = (float*)(w + 228996096ull);
  float* stats    = (float*)(w + 229124096ull);

  k_cvt<<<15625, 256, 0, stream>>>(Vs, lap, Vs16, lap16);
  k_temporal<<<1000, 256, 0, stream>>>(x, U1, U2, U3, part_lhs, part_rhs);
  k_E<<<1, 256, 0, stream>>>(part_lhs, part_rhs, be, Ve, E);
  k_pernode<<<16000, 256, 0, stream>>>(x, E, W1, W2, W3, tw, tb, xTAt, s_lhs, s_rhs, y_base);

  for (int b = 0; b < 4; b++) {
    size_t co = (size_t)b * 1024000;  // halves: 4000*256
    k_buildP<<<dim3(63, 63), 256, 0, stream>>>(s_lhs + b * 64000, s_rhs + b * 64000, bs, P16);
    gemm_wmma_f16<<<dim3(32, 32), 256, 0, stream>>>(Vs16, P16, Spre, nullptr, nullptr,
                                                    4000, 4000, 4000, 0);
    k_softmax_row<<<4000, 256, 0, stream>>>(Spre, S16);
    gemm_wmma_f16<<<dim3(2, 32), 256, 0, stream>>>(S16, xTAt + co, nullptr, Tx0 + co, nullptr,
                                                   4000, 4000, 256, 1);
    gemm_wmma_f16<<<dim3(2, 32), 256, 0, stream>>>(lap16, Tx0 + co, nullptr, Tx1 + co, nullptr,
                                                   4000, 4000, 256, 1);
    gemm_wmma_f16<<<dim3(2, 32), 256, 0, stream>>>(lap16, Tx1 + co, nullptr, Tx2 + co, Tx0 + co,
                                                   4000, 4000, 256, 2);
  }

  k_cheb<<<16000, 256, 0, stream>>>(Tx0, Tx1, Tx2, cheb, y_base, y_pre, part_st);
  k_stats<<<4, 256, 0, stream>>>(part_st, stats);
  k_norm<<<1000, 256, 0, stream>>>(y_pre, stats, ln_g, ln_b, out);

  (void)in_sizes; (void)n_in; (void)out_size; (void)ws_size;
}
